// sLSTMLayer_59657095741681
// MI455X (gfx1250) — compile-verified, matrix operand order checked by reading, not run on hardware
//
#include <hip/hip_runtime.h>

// ---------------- problem constants ----------------
#define BB   4
#define SS   2048
#define DD   1024
#define NH   4
#define DH   256

typedef __attribute__((ext_vector_type(16))) __bf16 v16bf;
typedef __attribute__((ext_vector_type(8)))  float  v8f;

// ---------------- async global->LDS (ASYNCcnt) support, probe-guarded ----------------
#if __has_builtin(__builtin_amdgcn_global_load_async_to_lds_b128)
#define HAVE_ASYNC 1
#else
#define HAVE_ASYNC 0
#endif

#if HAVE_ASYNC
#if __has_builtin(__builtin_amdgcn_s_wait_asynccnt)
#define WAIT_ASYNC(n) __builtin_amdgcn_s_wait_asynccnt(n)
#else
#define WAIT_ASYNC(n) asm volatile("s_wait_asynccnt %0" ::"i"(n) : "memory")
#endif
typedef int b128i __attribute__((vector_size(16)));          // matches builtin param pointee
typedef __attribute__((address_space(1))) b128i* gas_b128;   // global
typedef __attribute__((address_space(3))) b128i* las_b128;   // LDS
__device__ __forceinline__ void async_copy_b128(const __bf16* g, __bf16* l) {
    // AS casts via integers: flat LDS addr low 32 bits == LDS offset (ISA aperture rules)
    __builtin_amdgcn_global_load_async_to_lds_b128(
        (gas_b128)(unsigned long long)g,
        (las_b128)(unsigned int)(unsigned long long)l,
        0, 0);
}
#else
#define WAIT_ASYNC(n) ((void)0)
__device__ __forceinline__ void async_copy_b128(const __bf16* g, __bf16* l) {
    *(uint4*)l = *(const uint4*)g;   // synchronous fallback
}
#endif

__device__ __forceinline__ v8f wmma_bf16(v16bf a, v16bf b, v8f c) {
    return __builtin_amdgcn_wmma_f32_16x16x32_bf16(false, a, false, b, (short)0, c, false, false);
}

// K index inside a 16x32 bf16 A/B fragment for (lane, element e) per ISA 7.12.2
__device__ __forceinline__ int frag_k(int lane, int e) {
    int v = e >> 1;
    return ((v & 4) << 2) + ((lane & 16) >> 1) + ((v & 3) << 1) + (e & 1);
}

__device__ __forceinline__ v16bf frag_from_u4(uint4 lo, uint4 hi) {
    union { uint4 u[2]; v16bf v; } t;
    t.u[0] = lo; t.u[1] = hi;
    return t.v;
}

// ---------------- kernel 1: causal depthwise conv + swish, emit bf16 operands ----------------
__global__ __launch_bounds__(256) void conv_swish_kernel(
    const float* __restrict__ x, const float* __restrict__ cw, const float* __restrict__ cb,
    __bf16* __restrict__ xconv_bf, __bf16* __restrict__ x_bf)
{
    size_t idx = (size_t)blockIdx.x * 256 + threadIdx.x;   // over B*S*D
    int d = (int)(idx & (DD - 1));
    size_t sd = idx >> 10;
    int s = (int)(sd & (SS - 1));
    float xv = x[idx];
    float acc = cb[d] + xv * cw[d * 4 + 3];           // tap k multiplies x[s-3+k]
    if (s >= 1) acc += x[idx - DD]     * cw[d * 4 + 2];
    if (s >= 2) acc += x[idx - 2 * DD] * cw[d * 4 + 1];
    if (s >= 3) acc += x[idx - 3 * DD] * cw[d * 4 + 0];
    float sw = acc / (1.0f + __expf(-acc));           // swish
    xconv_bf[idx] = (__bf16)sw;
    x_bf[idx]     = (__bf16)xv;
}

// ---------------- kernel 2: weight conversion to WMMA-fragment-ordered bf16 ----------------
// wfrag: [g][h][kt(8)][nt(16)][lane(32)][e(16)]  value = w_g[h][n][k], k=kt*32+frag_k, n=nt*16+(lane&15)
// Rfrag: [h][kt(8)][nt(64)][lane(32)][e(16)]     value = rec[h][k][n]
__global__ __launch_bounds__(256) void cvt_weights_kernel(
    const float* __restrict__ w_i, const float* __restrict__ w_f,
    const float* __restrict__ w_z, const float* __restrict__ w_o,
    const float* __restrict__ rec,
    __bf16* __restrict__ wfrag, __bf16* __restrict__ Rfrag)
{
    size_t e = (size_t)blockIdx.x * 256 + threadIdx.x;
    const size_t WELEMS = (size_t)1 << 20;             // 4*4*8*16*32*16
    if (e < WELEMS) {
        int g    = (int)(e >> 18);
        int h    = (int)((e >> 16) & 3);
        int kt   = (int)((e >> 13) & 7);
        int nt   = (int)((e >> 9) & 15);
        int lane = (int)((e >> 4) & 31);
        int el   = (int)(e & 15);
        int k = kt * 32 + frag_k(lane, el);
        int n = nt * 16 + (lane & 15);
        const float* wp = (g == 0) ? w_i : (g == 1) ? w_f : (g == 2) ? w_z : w_o;
        wfrag[e] = (__bf16)wp[((size_t)h * DH + n) * DH + k];
    } else {
        size_t e2 = e - WELEMS;                        // < nh*dh*4dh = 2^20
        int h    = (int)(e2 >> 18);
        int kt   = (int)((e2 >> 15) & 7);
        int nt   = (int)((e2 >> 9) & 63);
        int lane = (int)((e2 >> 4) & 31);
        int el   = (int)(e2 & 15);
        int k = kt * 32 + frag_k(lane, el);
        int n = nt * 16 + (lane & 15);
        Rfrag[e2] = (__bf16)rec[((size_t)h * DH + k) * (4 * DH) + n];
    }
}

// ---------------- kernel 3: gate projection GEMM via WMMA + async double-buffered staging ----
// grid: (mtile=128, h=4, g=4); block 256 (8 waves). Tile M=64, N=256.
// wave w: mi = w>>1 (M-subtile), nb = w&1 (8 N-subtiles).
__global__ __launch_bounds__(256) void proj_gemm_kernel(
    const __bf16* __restrict__ xc_bf, const __bf16* __restrict__ x_bf,
    const __bf16* __restrict__ wfrag, float* __restrict__ gates)
{
    __shared__ __bf16 AsBuf[2][64 * 32];        // 4 KB x2
    __shared__ __bf16 BsBuf[2][16 * 32 * 16];   // 16 KB x2 (fragment-ordered)

    const int t    = threadIdx.x;
    const int lane = t & 31;
    const int w    = t >> 5;
    const int mi   = w >> 1;
    const int nb   = w & 1;
    const int m0   = blockIdx.x * 64;
    const int h    = blockIdx.y;
    const int g    = blockIdx.z;

    const __bf16* Asrc  = ((g < 2) ? xc_bf : x_bf) + (size_t)m0 * DD + h * DH;
    const __bf16* Wbase = wfrag + ((size_t)(g * NH + h) << 16);

    v8f acc[8];
#pragma unroll
    for (int j = 0; j < 8; ++j)
#pragma unroll
        for (int e = 0; e < 8; ++e) acc[j][e] = 0.0f;

    const int arow = t >> 2, aseg = t & 3;

    // ---- issue stage 0 ----
    {
        async_copy_b128(Asrc + (size_t)arow * DD + aseg * 8, &AsBuf[0][arow * 32 + aseg * 8]);
        const __bf16* Wkt = Wbase;                      // kt = 0
#pragma unroll
        for (int j = 0; j < 4; ++j) {
            int c = (t + j * 256) * 8;
            async_copy_b128(Wkt + c, &BsBuf[0][c]);
        }
    }

    for (int kt = 0; kt < 8; ++kt) {
        const int cur = kt & 1;
        if (kt < 7) {                                   // ---- issue stage kt+1 ----
            const int nxt = cur ^ 1;
            async_copy_b128(Asrc + (size_t)arow * DD + (kt + 1) * 32 + aseg * 8,
                            &AsBuf[nxt][arow * 32 + aseg * 8]);
            const __bf16* Wkt = Wbase + ((size_t)(kt + 1) << 13);
#pragma unroll
            for (int j = 0; j < 4; ++j) {
                int c = (t + j * 256) * 8;
                async_copy_b128(Wkt + c, &BsBuf[nxt][c]);
            }
            WAIT_ASYNC(5);                              // stage kt complete, kt+1 in flight
        } else {
            WAIT_ASYNC(0);
        }
        __syncthreads();

        // A fragment: two contiguous b128 LDS loads per lane
        const __bf16* Ab = &AsBuf[cur][(mi * 16 + (lane & 15)) * 32];
        v16bf a = frag_from_u4(*(const uint4*)(Ab + ((lane & 16) >> 1)),
                               *(const uint4*)(Ab + 16 + ((lane & 16) >> 1)));
#pragma unroll
        for (int j = 0; j < 8; ++j) {
            const __bf16* Bb = &BsBuf[cur][((nb * 8 + j) * 32 + lane) * 16];
            v16bf bf = frag_from_u4(*(const uint4*)Bb, *(const uint4*)(Bb + 8));
            acc[j] = wmma_bf16(a, bf, acc[j]);
        }
        __syncthreads();
    }

    // writeback: lanes 0-15: M=r, lanes 16-31: M=r+8; N = lane&15
#pragma unroll
    for (int j = 0; j < 8; ++j) {
#pragma unroll
        for (int r = 0; r < 8; ++r) {
            int m = m0 + mi * 16 + r + ((lane >> 4) << 3);   // global row = b*S + s
            int n = nb * 128 + j * 16 + (lane & 15);
            gates[((size_t)m * 4 + g) * DD + h * DH + n] = acc[j][r];
        }
    }
}

// ---------------- kernel 4: persistent sLSTM scan ----------------
// grid: 4 workgroups (one per head), 1024 threads (32 waves).
// R register-resident (fragment-ordered, two b128 loads per fragment).
__global__ __launch_bounds__(1024) void slstm_scan_kernel(
    const float* __restrict__ gates, const __bf16* __restrict__ Rfrag,
    const float* __restrict__ cbias, float* __restrict__ yv)
{
    __shared__ __bf16 h_lds[16][DH];      // 8 KB
    __shared__ float  rec_lds[4][4 * DH]; // 16 KB

    const int tid  = threadIdx.x;
    const int lane = tid & 31;
    const int w    = tid >> 5;
    const int h    = blockIdx.x;
    const int n0   = w * 32;

    // zero h state (also the M-padding rows 4..15)
#pragma unroll
    for (int j = 0; j < 4; ++j) ((__bf16*)h_lds)[tid + j * 1024] = (__bf16)0.0f;

    // preload R fragments: wave w owns N-subtiles {w*2, w*2+1}, all 8 K-subtiles
    const __bf16* Rh = Rfrag + ((size_t)h << 18);
    v16bf bfr[2][8];
#pragma unroll
    for (int tt = 0; tt < 2; ++tt)
#pragma unroll
        for (int kt = 0; kt < 8; ++kt) {
            const __bf16* p = Rh + ((size_t)((kt * 64 + (w * 2 + tt)) * 32 + lane) * 16);
            bfr[tt][kt] = frag_from_u4(*(const uint4*)p, *(const uint4*)(p + 8));
        }

    // per-thread pointwise state: thread (b, d)
    const int b = tid >> 8;
    const int d = tid & 255;
    float c_s = 0.0f, n_s = 0.0f, m_s = 0.0f;
    const float bi  = cbias[(0 * NH + h) * DH + d];
    const float bfv = cbias[(1 * NH + h) * DH + d];
    const float bz  = cbias[(2 * NH + h) * DH + d];
    const float bo  = cbias[(3 * NH + h) * DH + d];
    const float* grow = gates + ((size_t)b * SS * 4) * DD + h * DH + d;
    float* yrow = yv + ((size_t)(b * NH + h) * SS) * DH + d;

    __syncthreads();

    for (int s = 0; s < SS; ++s) {
        // ---- phase 1: rec = h x R via WMMA ----
        v8f acc0, acc1;
#pragma unroll
        for (int e = 0; e < 8; ++e) { acc0[e] = 0.0f; acc1[e] = 0.0f; }
        const __bf16* hr = &h_lds[lane & 15][0];
#pragma unroll
        for (int kt = 0; kt < 8; ++kt) {
            v16bf a = frag_from_u4(*(const uint4*)(hr + kt * 32 + ((lane & 16) >> 1)),
                                   *(const uint4*)(hr + kt * 32 + 16 + ((lane & 16) >> 1)));
            acc0 = wmma_bf16(a, bfr[0][kt], acc0);
            acc1 = wmma_bf16(a, bfr[1][kt], acc1);
        }
#pragma unroll
        for (int r = 0; r < 4; ++r) {
            if (lane < 16) {
                rec_lds[r][n0 + (lane & 15)]      = acc0[r];
                rec_lds[r][n0 + 16 + (lane & 15)] = acc1[r];
            }
        }
        __syncthreads();

        // ---- phase 2: stabilized gate update ----
        const float* gp = grow + (size_t)s * (4 * DD);
        if (s + 1 < SS) __builtin_prefetch(gp + 4 * DD, 0, 0);   // global_prefetch_b8

        float iraw = gp[0 * DD] + rec_lds[b][0 * DH + d] + bi;
        float fraw = gp[1 * DD] + rec_lds[b][1 * DH + d] + bfv;
        float zraw = gp[2 * DD] + rec_lds[b][2 * DH + d] + bz;
        float oraw = gp[3 * DD] + rec_lds[b][3 * DH + d] + bo;

        float logsigf = fminf(fraw, 0.0f) - log1pf(__expf(-fabsf(fraw)));
        float lfm     = m_s + logsigf;
        float m_new   = fmaxf(iraw, lfm);
        float ig      = __expf(iraw - m_new);
        float fg      = __expf(lfm - m_new);
        c_s = fg * c_s + ig * tanhf(zraw);
        n_s = fg * n_s + ig;
        m_s = m_new;
        float sig_o = 1.0f / (1.0f + __expf(-oraw));
        float h_new = sig_o * c_s / n_s;

        yrow[(size_t)s * DH] = h_new;
        h_lds[b][d] = (__bf16)h_new;
        __syncthreads();
    }
}

// ---------------- kernel 5: multihead layernorm + output transpose ----------------
__global__ __launch_bounds__(256) void mh_layernorm_kernel(
    const float* __restrict__ yv, const float* __restrict__ scale,
    const float* __restrict__ bias, float* __restrict__ out)
{
    const int w    = threadIdx.x >> 5;
    const int lane = threadIdx.x & 31;
    const int row  = blockIdx.x * 8 + w;        // (b*NH + h)*S + s
    const int s    = row & (SS - 1);
    const int bh   = row >> 11;
    const int b    = bh >> 2;
    const int h    = bh & 3;

    const float* src = yv + (size_t)row * DH;
    float vals[8], sum = 0.0f, sq = 0.0f;
#pragma unroll
    for (int j = 0; j < 8; ++j) {
        float v = src[lane + j * 32];
        vals[j] = v; sum += v; sq += v * v;
    }
#pragma unroll
    for (int mk = 16; mk; mk >>= 1) {
        sum += __shfl_xor(sum, mk, 32);
        sq  += __shfl_xor(sq,  mk, 32);
    }
    float mu   = sum * (1.0f / DH);
    float var  = sq * (1.0f / DH) - mu * mu;
    float rinv = rsqrtf(var + 1e-6f);

    float* dst = out + ((size_t)(b * SS + s)) * DD + h * DH;
#pragma unroll
    for (int j = 0; j < 8; ++j) {
        int dd = lane + j * 32;
        dst[dd] = (vals[j] - mu) * rinv * scale[h * DH + dd] + bias[h * DH + dd];
    }
}

// ---------------- host launcher ----------------
extern "C" void kernel_launch(void* const* d_in, const int* in_sizes, int n_in,
                              void* d_out, int out_size, void* d_ws, size_t ws_size,
                              hipStream_t stream) {
    const float* x      = (const float*)d_in[0];
    const float* conv_w = (const float*)d_in[1];
    const float* conv_b = (const float*)d_in[2];
    const float* w_i    = (const float*)d_in[3];
    const float* w_f    = (const float*)d_in[4];
    const float* w_z    = (const float*)d_in[5];
    const float* w_o    = (const float*)d_in[6];
    const float* rec    = (const float*)d_in[7];
    const float* cbias  = (const float*)d_in[8];
    const float* gsc    = (const float*)d_in[9];
    const float* gbi    = (const float*)d_in[10];
    float* out = (float*)d_out;

    const size_t NXD = (size_t)BB * SS * DD;              // 8M elems
    __bf16* xc_bf  = (__bf16*)d_ws;
    __bf16* x_bf   = xc_bf + NXD;
    __bf16* wfrag  = x_bf + NXD;                          // 2^20 elems
    __bf16* Rfrag  = wfrag + ((size_t)1 << 20);           // 2^20 elems
    float*  gates  = (float*)(Rfrag + ((size_t)1 << 20));
    float*  yv     = gates + (size_t)BB * SS * 4 * DD;    // 32M floats

    conv_swish_kernel<<<(unsigned)(NXD / 256), 256, 0, stream>>>(x, conv_w, conv_b, xc_bf, x_bf);
    cvt_weights_kernel<<<(2 * 1048576) / 256, 256, 0, stream>>>(w_i, w_f, w_z, w_o, rec, wfrag, Rfrag);
    proj_gemm_kernel<<<dim3(BB * SS / 64, NH, 4), 256, 0, stream>>>(xc_bf, x_bf, wfrag, gates);
    slstm_scan_kernel<<<NH, 1024, 0, stream>>>(gates, Rfrag, cbias, yv);
    mh_layernorm_kernel<<<BB * NH * SS / 8, 256, 0, stream>>>(yv, gsc, gbi, out);
}
